// ActiveParticles_55173149884894
// MI455X (gfx1250) — compile-verified, matrix operand order checked by reading, not run on hardware
//
#include <hip/hip_runtime.h>
#include <hip/hip_bf16.h>

#define N_PART 4096
#define CHUNK  1024

typedef _Float16 v16h __attribute__((ext_vector_type(16)));
typedef _Float16 h8   __attribute__((ext_vector_type(8)));
typedef float    v8f  __attribute__((ext_vector_type(8)));

// ---- physical constants (from reference) ----
__device__ __constant__ float kCenter = 0.64e-3f;                 // L/2
#define K_RC      3.15e-6f
#define K_RR2     (8e-6f * 8e-6f)                                  // RR^2
#define K_RO2     ((2.5e-5f + 3.15e-6f) * (2.5e-5f + 3.15e-6f))    // (RO+RC)^2
#define K_COLL2   ((2.0f * 3.15e-6f) * (2.0f * 3.15e-6f))          // (2 RC)^2
#define K_SDT     0.44721359549995794f                             // sqrt(DT=0.2)
#define K_S2DR    0.07483314773547883f                             // sqrt(2*DR)
#define K_ROTGAIN 0.014f                                           // DT*GAMMA*DR
#define K_DTVEL   1.0e-7f                                          // DT*VEL
#define K_TN      (0.7071067811865476f * 1.6733200530681511e-7f)   // sqrt(.5)*sqrt(2*DT_TRANS)

// ============ kernel 1: global mean of positions (Wra = all ones) ============
__global__ __launch_bounds__(256) void apMean(const float* __restrict__ pre,
                                              const float* __restrict__ pim,
                                              float* __restrict__ ws) {
    __shared__ float sx[256], sy[256];
    const int tid = threadIdx.x;
    float ax = 0.f, ay = 0.f;
    for (int j = tid; j < N_PART; j += 256) { ax += pre[j]; ay += pim[j]; }
    sx[tid] = ax; sy[tid] = ay;
    __syncthreads();
    for (int s = 128; s > 0; s >>= 1) {
        if (tid < s) { sx[tid] += sx[tid + s]; sy[tid] += sy[tid + s]; }
        __syncthreads();
    }
    if (tid == 0) {
        ws[0] = sx[0] * (1.0f / N_PART) - kCenter;  // centered mean
        ws[1] = sy[0] * (1.0f / N_PART) - kCenter;
    }
}

// ============ kernel 2: forward pass with WMMA masked reductions ============
// D = A x B.  A (16x32 f16): rows 0-7 = {pre_hi,pre_lo,pim_hi,pim_lo,
// ore_hi,ore_lo,oim_hi,oim_lo}, row 8 = ones (free neighbor count), rows 9-15 = 0.
// K = 32 particles j.  B (32x16 f16) = mask[j][i] for 16 output particles i.
// Self pair (d2 == 0) passes both masks; its exact contribution is removed
// at finalize (Wrr) or wanted anyway (Wro includes the diagonal).
__global__ __launch_bounds__(256) void apForward(
    const float* __restrict__ pre, const float* __restrict__ pim,
    const float* __restrict__ ore, const float* __restrict__ oim,
    const float* __restrict__ deltas, const float* __restrict__ rnoise,
    const float* __restrict__ tnr, const float* __restrict__ tni,
    const float* __restrict__ ws, float2* __restrict__ posOut,
    float* __restrict__ out) {
    __shared__ _Float16 vecs[16 * CHUNK];  // 32 KB: 16 A-rows
    __shared__ float4   jd[CHUNK];         // 16 KB: (pre_c, pim_c, ore, oim)

    const int tid    = threadIdx.x;
    const int lane   = tid & 31;
    const int wave   = tid >> 5;
    const int li     = lane & 15;          // A row m / B output column i
    const int hiHalf = lane >> 4;          // K-group select
    const int ibase  = (blockIdx.x * 8 + wave) * 16;
    const int i      = ibase + li;

    // constant A rows 8..15 (ones row + zero padding), written once
    for (int jj = tid; jj < CHUNK; jj += 256) {
        vecs[8 * CHUNK + jj] = (_Float16)1.f;
#pragma unroll
        for (int m = 9; m < 16; ++m) vecs[m * CHUNK + jj] = (_Float16)0.f;
    }

    const float mex = pre[i] - kCenter;
    const float mey = pim[i] - kCenter;
    const float moz = ore[i];
    const float mow = oim[i];

    v8f drr = {0.f,0.f,0.f,0.f,0.f,0.f,0.f,0.f};
    v8f dro = {0.f,0.f,0.f,0.f,0.f,0.f,0.f,0.f};

    for (int c0 = 0; c0 < N_PART; c0 += CHUNK) {
        __syncthreads();
        for (int jj = tid; jj < CHUNK; jj += 256) {
            const int j = c0 + jj;
            const float px = pre[j] - kCenter;
            const float py = pim[j] - kCenter;
            const float ox = ore[j];
            const float oy = oim[j];
            const _Float16 ah = (_Float16)px; const _Float16 al = (_Float16)(px - (float)ah);
            const _Float16 bh = (_Float16)py; const _Float16 bl = (_Float16)(py - (float)bh);
            const _Float16 ch = (_Float16)ox; const _Float16 cl = (_Float16)(ox - (float)ch);
            const _Float16 dh = (_Float16)oy; const _Float16 dl = (_Float16)(oy - (float)dh);
            vecs[0*CHUNK + jj] = ah; vecs[1*CHUNK + jj] = al;
            vecs[2*CHUNK + jj] = bh; vecs[3*CHUNK + jj] = bl;
            vecs[4*CHUNK + jj] = ch; vecs[5*CHUNK + jj] = cl;
            vecs[6*CHUNK + jj] = dh; vecs[7*CHUNK + jj] = dl;
            jd[jj] = make_float4(px, py, ox, oy);
        }
        __syncthreads();

        for (int kb = 0; kb < CHUNK / 32; ++kb) {
            const int jbl = kb * 32;
            const int j0l = jbl + hiHalf * 8;  // local K indices 0..7 (or 8..15)
            const int j1l = j0l + 16;          // local K indices 16..23 (or 24..31)

            // unconditional A-fragment loads (rows 8..15 pre-filled)
            const h8 alo = *(const h8*)(vecs + li * CHUNK + j0l);
            const h8 ahi = *(const h8*)(vecs + li * CHUNK + j1l);
            const v16h A = __builtin_shufflevector(alo, ahi,
                                                   0,1,2,3,4,5,6,7,8,9,10,11,12,13,14,15);

            v16h Brr, Bro;
#pragma unroll
            for (int t = 0; t < 16; ++t) {
                const int jl = (t < 8) ? (j0l + t) : (j1l + t - 8);
                const float4 pj = jd[jl];
                const float dx = pj.x - mex;
                const float dy = pj.y - mey;
                const float d2 = dx * dx + dy * dy;
                const float dp = pj.z * moz + pj.w * mow;     // ori_i . ori_j
                const bool mrr = (d2 <= K_RR2) & (dp > 0.f);  // self passes; fixed later
                const bool mro = (d2 <= K_RO2);               // self passes (wanted)
                Brr[t] = mrr ? (_Float16)1.f : (_Float16)0.f;
                Bro[t] = mro ? (_Float16)1.f : (_Float16)0.f;
            }

            drr = __builtin_amdgcn_wmma_f32_16x16x32_f16(false, A, false, Brr,
                                                         (short)0, drr, false, false);
            dro = __builtin_amdgcn_wmma_f32_16x16x32_f16(false, A, false, Bro,
                                                         (short)0, dro, false, false);
        }
    }

    // D row 8 (= Sum Wrr * ones) lives in lanes 16-31, VGPR0, col = lane-16;
    // subtract the self pair to get n_r.
    const float nr = __shfl(drr[0], li + 16, 32) - 1.0f;

    if (lane < 16) {
        const float px = pre[i], py = pim[i];
        const float ox = ore[i], oy = oim[i];
        const float pcx = px - kCenter, pcy = py - kCenter;

        // hi+lo recombination; remove self contribution from Wrr sums
        const float sre = (drr[0] + drr[1]) - pcx;   // sum_{j!=i} Wrr * pos_re (centered)
        const float sim = (drr[2] + drr[3]) - pcy;   // sum_{j!=i} Wrr * pos_im (centered)
        const float osr = dro[4] + dro[5];           // sum Wro * ori_re (self included)
        const float osi = dro[6] + dro[7];           // sum Wro * ori_im

        const float sgn = (nr > 0.f) ? 1.f : 0.f;
        const float inv = 1.f / fmaxf(nr, 1.f);
        const float Sx = sre * inv - pcx * sgn;      // centering cancels exactly
        const float Sy = sim * inv - pcy * sgn;
        const float dxv = -Sx, dyv = -Sy;            // d = -S

        const float Px = ws[0] - pcx;                // Ps = mean - pos
        const float Py = ws[1] - pcy;

        float sd, cd;
        __sincosf(deltas[i], &sd, &cd);              // args in [0, 0.5): hw-accurate
        const float lx = Px * cd - Py * sd, ly = Px * sd + Py * cd;   // Ps*e^{+i d}
        const float rx = Px * cd + Py * sd, ry = Py * cd - Px * sd;   // Ps*e^{-i d}

        // cossim compare reduces to dot compare (equal norms)
        const bool useL = (lx * osr + ly * osi) >= (rx * osr + ry * osi);
        const float bx = useL ? lx : rx;
        const float by = useL ? ly : ry;

        const bool hasRep = (dxv != 0.f) || (dyv != 0.f);
        const float zx = hasRep ? dxv : bx;
        const float zy = hasRep ? dyv : by;

        // sin(att) = sin(angle(z) - angle(ori)) = (zy*ox - zx*oy)/(|z||ori|)
        const float az = sqrtf(zx * zx + zy * zy);
        const float ao = sqrtf(ox * ox + oy * oy);
        const float sinatt = (az > 0.f) ? (zy * ox - zx * oy) / (az * ao)
                                        : (-oy / ao);    // angle(0)=0 path

        const float theta = K_ROTGAIN * sinatt + rnoise[i] * K_S2DR * K_SDT;
        float sr, cr;
        __sincosf(theta, &sr, &cr);                  // |theta| < ~0.2: hw-accurate
        const float nox = ox * cr - oy * sr;
        const float noy = ox * sr + oy * cr;

        const float tx = K_DTVEL * ox + tnr[i] * K_TN * K_SDT;
        const float ty = K_DTVEL * oy + tni[i] * K_TN * K_SDT;

        posOut[i] = make_float2(px + tx, py + ty);   // pre-collision position

        out[(1 * N_PART + i) * 2 + 0] = nox;  out[(1 * N_PART + i) * 2 + 1] = noy;
        out[(2 * N_PART + i) * 2 + 0] = osr;  out[(2 * N_PART + i) * 2 + 1] = osi;
        out[(3 * N_PART + i) * 2 + 0] = lx;   out[(3 * N_PART + i) * 2 + 1] = ly;
        out[(4 * N_PART + i) * 2 + 0] = rx;   out[(4 * N_PART + i) * 2 + 1] = ry;
    }
}

// ============ kernel 3: one synchronous collision-resolution iteration ============
__global__ __launch_bounds__(256) void apCollide(const float2* __restrict__ pin,
                                                 float2* __restrict__ pout,
                                                 float* __restrict__ finalOut) {
    __shared__ float2 sj[256];
    const int i = blockIdx.x * 256 + threadIdx.x;
    const float2 p = pin[i];
    float ax = 0.f, ay = 0.f;
    for (int c = 0; c < N_PART / 256; ++c) {
        __syncthreads();
        sj[threadIdx.x] = pin[c * 256 + threadIdx.x];
        __syncthreads();
        for (int t = 0; t < 256; ++t) {
            const int j = c * 256 + t;
            const float dx = sj[t].x - p.x;
            const float dy = sj[t].y - p.y;
            const float d2 = dx * dx + dy * dy;
            if ((d2 <= K_COLL2) && (j != i)) {
                const float ad = sqrtf(d2);
                const float sc = (2.1f * K_RC - ad) * 0.5f / ad;
                ax = fmaf(dx, sc, ax);
                ay = fmaf(dy, sc, ay);
            }
        }
    }
    const float nx = p.x - ax;
    const float ny = p.y - ay;
    pout[i] = make_float2(nx, ny);
    if (finalOut) { finalOut[2 * i] = nx; finalOut[2 * i + 1] = ny; }
}

// ============ host launcher ============
extern "C" void kernel_launch(void* const* d_in, const int* in_sizes, int n_in,
                              void* d_out, int out_size, void* d_ws, size_t ws_size,
                              hipStream_t stream) {
    (void)in_sizes; (void)n_in; (void)out_size; (void)ws_size;
    const float* pre = (const float*)d_in[0];
    const float* pim = (const float*)d_in[1];
    const float* ore = (const float*)d_in[2];
    const float* oim = (const float*)d_in[3];
    const float* del = (const float*)d_in[4];
    const float* rno = (const float*)d_in[5];
    const float* tnr = (const float*)d_in[6];
    const float* tni = (const float*)d_in[7];
    float* out = (float*)d_out;
    float* ws  = (float*)d_ws;
    float2* posA = (float2*)(ws + 8);
    float2* posB = posA + N_PART;

    apMean<<<1, 256, 0, stream>>>(pre, pim, ws);
    apForward<<<32, 256, 0, stream>>>(pre, pim, ore, oim, del, rno, tnr, tni,
                                      ws, posA, out);

    const float2* src = posA;
    float2* dst = posB;
    for (int it = 0; it < 30; ++it) {
        float* fin = (it == 29) ? out : (float*)nullptr;   // new_pos = output block 0
        apCollide<<<16, 256, 0, stream>>>(src, dst, fin);
        const float2* tmp = dst;
        dst = (float2*)src;
        src = tmp;
    }
}
